// GCMCGraphConv_77077483094300
// MI455X (gfx1250) — matrix operand order repaired and checked by reading, not compiled, not run on hardware
//
#include <hip/hip_runtime.h>
#include <hip/hip_bf16.h>
#include <math.h>

typedef _Float16 f16;
typedef __attribute__((ext_vector_type(8)))  _Float16 v8h;
typedef __attribute__((ext_vector_type(16))) _Float16 v16h;
typedef __attribute__((ext_vector_type(8)))  float    v8f;

#define FEAT 256
#define RDIM 128
#define ET   32          // edges per block in the fused edge kernel
#define S128 136         // padded LDS stride (f16 elems) for 128-wide tiles
#define S256 264         // padded LDS stride (f16 elems) for 256-wide tiles

// Branch-free tanh-form GELU: x * sigmoid(2*sqrt(2/pi)*(x + 0.044715 x^3))
// exp folded to base-2 (v_exp_f32), rcp via v_rcp_f32. Limits are exact at +-inf.
__device__ __forceinline__ float gelu_fast(float x) {
    const float a = 2.30220795f;   // 2*sqrt(2/pi)/ln2
    const float b = 0.10294387f;   // 2*sqrt(2/pi)*0.044715/ln2
    float x2 = x * x;
    float t  = x * __builtin_fmaf(b, x2, a);          // exponent (base-2) argument
    float e  = __builtin_amdgcn_exp2f(-t);            // 2^-t = exp(-2u)
    return x * __builtin_amdgcn_rcpf(1.0f + e);       // x * sigmoid(2u)
}
__device__ __forceinline__ float sigm(float x) {
    float e = __builtin_amdgcn_exp2f(-1.44269504f * x);
    return __builtin_amdgcn_rcpf(1.0f + e);
}

// A fragment: 16x32 f16 tile from LDS (row-major, padded stride).
// Lane l: M=l&15, hi=l>>4. v16h elems 0..7 = K[hi*8 .. hi*8+7] (contiguous, 16B),
// elems 8..15 = K[16+hi*8 ..] (contiguous, 16B) -> two ds_load_b128.
template <int STRIDE>
__device__ __forceinline__ v16h load_a_frag(const f16* sA, int rowBase, int kBase, int lane) {
    const int m  = lane & 15;
    const int hi = lane >> 4;
    const f16* rp = sA + (rowBase + m) * STRIDE + kBase + hi * 8;
    v8h lo = *(const v8h*)(rp);
    v8h hh = *(const v8h*)(rp + 16);
    return __builtin_shufflevector(lo, hh, 0, 1, 2, 3, 4, 5, 6, 7,
                                           8, 9, 10, 11, 12, 13, 14, 15);
}

// B fragment: 32x16 f16 (KxN) from row-major weight W[n][k] (B[k][n] = W[n][k]).
// Lane l: N=l&15, hi=l>>4; elems 0..7 = K[hi*8..], elems 8..15 = K[16+hi*8..] (two 16B loads).
__device__ __forceinline__ v16h load_b_frag(const f16* W, int K, int colBase, int kBase, int lane) {
    const int n  = lane & 15;
    const int hi = lane >> 4;
    const f16* rp = W + (long)(colBase + n) * K + kBase + hi * 8;
    v8h lo = *(const v8h*)(rp);
    v8h hh = *(const v8h*)(rp + 16);
    return __builtin_shufflevector(lo, hh, 0, 1, 2, 3, 4, 5, 6, 7,
                                           8, 9, 10, 11, 12, 13, 14, 15);
}

// ---------------------------------------------------------------- utilities
__global__ void cvt_f32_to_f16(const float* __restrict__ src, f16* __restrict__ dst, int n) {
    int i = blockIdx.x * blockDim.x + threadIdx.x;
    int stride = gridDim.x * blockDim.x;
    for (; i < n; i += stride) dst[i] = (f16)src[i];
}

__global__ void zero_f32v4(float4* __restrict__ p, int n4) {
    int i = blockIdx.x * blockDim.x + threadIdx.x;
    int stride = gridDim.x * blockDim.x;
    float4 z = make_float4(0.f, 0.f, 0.f, 0.f);
    for (; i < n4; i += stride) p[i] = z;
}

// ---------------------------------------------------------------- fused edge kernel
__global__ __launch_bounds__(256, 2) void edge_mlp_scatter(
    const float* __restrict__ feat, const float* __restrict__ cj,
    const int* __restrict__ esrc, const int* __restrict__ edst,
    const int* __restrict__ rid, const float* __restrict__ remb,
    const float* __restrict__ probw, const float* __restrict__ scorew,
    const f16* __restrict__ rw1h, const f16* __restrict__ rw2h, const f16* __restrict__ rw3h,
    float* __restrict__ h, int nE)
{
    __shared__ f16  sX1[ET * S256];   // layer-1 activations (f16)
    __shared__ f16  sB [ET * S256];   // rfeat (first ET*S128), later layer-2 activations
    __shared__ float s_pa[ET], s_ra[ET], s_cj[ET];
    __shared__ int   s_src[ET], s_dst[ET];

    const int tid  = threadIdx.x;
    const int lane = tid & 31;
    const int wave = tid >> 5;
    const int e0   = blockIdx.x * ET;
    f16* sR = sB;                      // rfeat region [ET][S128]

    // Phase 0a: gather review embeddings -> LDS f16 (zero-pad tail edges)
    for (int idx = tid; idx < ET * RDIM; idx += 256) {
        int r = idx >> 7, k = idx & (RDIM - 1);
        int e = e0 + r;
        float v = 0.0f;
        if (e < nE) v = remb[(long)rid[e] * RDIM + k];
        sR[r * S128 + k] = (f16)v;
    }
    __syncthreads();

    // Phase 0b: per-edge gates pa/ra and src-side cj / indices
    if (tid < ET) {
        int e = e0 + tid;
        float accp = 0.0f, accr = 0.0f;
        const f16* rp = sR + tid * S128;
#pragma unroll 4
        for (int k = 0; k < RDIM; ++k) {
            float v = (float)rp[k];
            accp += v * probw[k];
            accr += v * scorew[k];
        }
        bool ok = (e < nE);
        s_pa[tid]  = ok ? sigm(accp) : 0.0f;
        s_ra[tid]  = ok ? sigm(accr) : 0.0f;
        int se     = ok ? esrc[e] : 0;
        s_src[tid] = se;
        s_cj[tid]  = ok ? cj[se] : 0.0f;
        s_dst[tid] = ok ? edst[e] : 0;
    }
    __syncthreads();

    const int n  = lane & 15;
    const int hi = lane >> 4;

    // Phase 1: X1 = gelu(rfeat @ rw1^T)   [ET,128] x [128,256]
    for (int ct = wave * 2; ct < wave * 2 + 2; ++ct) {
        v16h b[4];
#pragma unroll
        for (int s = 0; s < 4; ++s) b[s] = load_b_frag(rw1h, RDIM, ct * 16, s * 32, lane);
#pragma unroll
        for (int rt = 0; rt < ET / 16; ++rt) {
            v8f c = {};
#pragma unroll
            for (int s = 0; s < 4; ++s) {
                v16h a = load_a_frag<S128>(sR, rt * 16, s * 32, lane);
                c = __builtin_amdgcn_wmma_f32_16x16x32_f16(false, a, false, b[s],
                                                           (short)0, c, false, false);
            }
#pragma unroll
            for (int vi = 0; vi < 8; ++vi) {
                int row = rt * 16 + vi + 8 * hi;
                sX1[row * S256 + ct * 16 + n] = (f16)gelu_fast(c[vi]);
            }
        }
    }
    __syncthreads();

    // Phase 2: X2 = gelu(X1 @ rw2^T)  -> overwrite sB (rfeat is dead)
    f16* sX2 = sB;
    for (int ct = wave * 2; ct < wave * 2 + 2; ++ct) {
        v16h b[8];
#pragma unroll
        for (int s = 0; s < 8; ++s) b[s] = load_b_frag(rw2h, FEAT, ct * 16, s * 32, lane);
#pragma unroll
        for (int rt = 0; rt < ET / 16; ++rt) {
            v8f c = {};
#pragma unroll
            for (int s = 0; s < 8; ++s) {
                v16h a = load_a_frag<S256>(sX1, rt * 16, s * 32, lane);
                c = __builtin_amdgcn_wmma_f32_16x16x32_f16(false, a, false, b[s],
                                                           (short)0, c, false, false);
            }
#pragma unroll
            for (int vi = 0; vi < 8; ++vi) {
                int row = rt * 16 + vi + 8 * hi;
                sX2[row * S256 + ct * 16 + n] = (f16)gelu_fast(c[vi]);
            }
        }
    }
    __syncthreads();

    // Phase 3: rf = X2 @ rw3^T, fused epilogue:
    //   m = (feat[src]*pa + rf*ra) * cj ; atomicAdd into h[dst]
    for (int ct = wave * 2; ct < wave * 2 + 2; ++ct) {
        v16h b[8];
#pragma unroll
        for (int s = 0; s < 8; ++s) b[s] = load_b_frag(rw3h, FEAT, ct * 16, s * 32, lane);
#pragma unroll
        for (int rt = 0; rt < ET / 16; ++rt) {
            v8f c = {};
#pragma unroll
            for (int s = 0; s < 8; ++s) {
                v16h a = load_a_frag<S256>(sX2, rt * 16, s * 32, lane);
                c = __builtin_amdgcn_wmma_f32_16x16x32_f16(false, a, false, b[s],
                                                           (short)0, c, false, false);
            }
#pragma unroll
            for (int vi = 0; vi < 8; ++vi) {
                int row = rt * 16 + vi + 8 * hi;
                int e   = e0 + row;
                if (e < nE) {
                    int col  = ct * 16 + n;
                    float fv = feat[(long)s_src[row] * FEAT + col];
                    float m  = (fv * s_pa[row] + c[vi] * s_ra[row]) * s_cj[row];
                    atomicAdd(&h[(long)s_dst[row] * FEAT + col], m);
                }
            }
        }
    }
}

// ---------------------------------------------------------------- output GEMM
// out = (h * ci) @ lin_w^T + lin_b, in-place over d_out (each block stages its
// own 16 rows of h into LDS before overwriting them).
__global__ __launch_bounds__(256, 2) void out_gemm(
    const float* __restrict__ ci, const f16* __restrict__ linwh,
    const float* __restrict__ linb, float* __restrict__ hout, int nD)
{
    __shared__ f16 sA[16 * S256];
    const int tid  = threadIdx.x;
    const int lane = tid & 31;
    const int wave = tid >> 5;
    const int r0   = blockIdx.x * 16;

    for (int idx = tid; idx < 16 * FEAT; idx += 256) {
        int r = idx >> 8, k = idx & (FEAT - 1);
        int row = r0 + r;
        float v = 0.0f;
        if (row < nD) v = hout[(long)row * FEAT + k] * ci[row];
        sA[r * S256 + k] = (f16)v;
    }
    __syncthreads();

    const int n  = lane & 15;
    const int hi = lane >> 4;
    for (int ct = wave * 2; ct < wave * 2 + 2; ++ct) {
        v16h b[8];
#pragma unroll
        for (int s = 0; s < 8; ++s) b[s] = load_b_frag(linwh, FEAT, ct * 16, s * 32, lane);
        v8f c = {};
#pragma unroll
        for (int s = 0; s < 8; ++s) {
            v16h a = load_a_frag<S256>(sA, 0, s * 32, lane);
            c = __builtin_amdgcn_wmma_f32_16x16x32_f16(false, a, false, b[s],
                                                       (short)0, c, false, false);
        }
#pragma unroll
        for (int vi = 0; vi < 8; ++vi) {
            int row = r0 + vi + 8 * hi;
            if (row < nD) {
                int col = ct * 16 + n;
                hout[(long)row * FEAT + col] = c[vi] + linb[col];
            }
        }
    }
}

// ---------------------------------------------------------------- launcher
extern "C" void kernel_launch(void* const* d_in, const int* in_sizes, int n_in,
                              void* d_out, int out_size, void* d_ws, size_t ws_size,
                              hipStream_t stream) {
    const float* feat   = (const float*)d_in[0];
    const float* cj     = (const float*)d_in[1];
    const float* ci     = (const float*)d_in[2];
    const int*   esrc   = (const int*)  d_in[3];
    const int*   edst   = (const int*)  d_in[4];
    const int*   rid    = (const int*)  d_in[5];
    const float* remb   = (const float*)d_in[6];
    const float* probw  = (const float*)d_in[7];
    const float* scorew = (const float*)d_in[8];
    const float* rw1    = (const float*)d_in[9];   // [256,128]
    const float* rw2    = (const float*)d_in[10];  // [256,256]
    const float* rw3    = (const float*)d_in[11];  // [256,256]
    const float* linw   = (const float*)d_in[12];  // [256,256]
    const float* linb   = (const float*)d_in[13];  // [256]

    const int nE = in_sizes[3];   // edges
    const int nD = in_sizes[2];   // destination nodes (ci)

    // f16 weight copies in workspace (64KB-aligned regions)
    char* ws    = (char*)d_ws;
    f16* rw1h   = (f16*)(ws);
    f16* rw2h   = (f16*)(ws + 1 * 65536 * 2);
    f16* rw3h   = (f16*)(ws + 3 * 65536 * 2);
    f16* linwh  = (f16*)(ws + 5 * 65536 * 2);

    cvt_f32_to_f16<<<128, 256, 0, stream>>>(rw1,  rw1h,  FEAT * RDIM);
    cvt_f32_to_f16<<<256, 256, 0, stream>>>(rw2,  rw2h,  FEAT * FEAT);
    cvt_f32_to_f16<<<256, 256, 0, stream>>>(rw3,  rw3h,  FEAT * FEAT);
    cvt_f32_to_f16<<<256, 256, 0, stream>>>(linw, linwh, FEAT * FEAT);

    float* h = (float*)d_out;                       // accumulator lives in d_out
    zero_f32v4<<<2048, 256, 0, stream>>>((float4*)h, out_size / 4);

    int nBlocksE = (nE + ET - 1) / ET;
    edge_mlp_scatter<<<nBlocksE, 256, 0, stream>>>(feat, cj, esrc, edst, rid, remb,
                                                   probw, scorew, rw1h, rw2h, rw3h,
                                                   h, nE);

    int nBlocksO = (nD + 15) / 16;
    out_gemm<<<nBlocksO, 256, 0, stream>>>(ci, linwh, linb, h, nD);
}
